// GPT2Attention_11510512353458
// MI455X (gfx1250) — matrix operand order repaired and checked by reading, not compiled, not run on hardware
//
#include <hip/hip_runtime.h>

// GPT-2 attention: B=2, S=2048, D=1024, H=16, HS=64
#define BB  2
#define SS  2048
#define DD  1024
#define HH  16
#define HSZ 64

typedef __bf16 bf16_t;
typedef __attribute__((ext_vector_type(16))) __bf16 v16bf;
typedef __attribute__((ext_vector_type(8)))  __bf16 v8bf;
typedef __attribute__((ext_vector_type(8)))  float  v8f;

union BF16Frag { v16bf v; v8bf h[2]; };

__device__ __forceinline__ bf16_t f2bf(float f) {
  return (bf16_t)f;                       // native fptrunc -> v_cvt bf16
}

__device__ __forceinline__ v8bf cvt8(v8f f) {
  return __builtin_convertvector(f, v8bf); // packed bf16 converts
}

__device__ __forceinline__ v8f wmma_bf16(const BF16Frag& a, const BF16Frag& b, v8f c) {
  // (neg_a, A, neg_b, B, c_mod, C, reuse_a, reuse_b)
  return __builtin_amdgcn_wmma_f32_16x16x32_bf16(false, a.v, false, b.v, (short)0, c, false, false);
}

__device__ __forceinline__ v8f v8f_zero() {
  v8f z;
#pragma unroll
  for (int j = 0; j < 8; ++j) z[j] = 0.0f;
  return z;
}

__device__ __forceinline__ v8f pack8(float4 a, float4 b) {
  v8f r;
  r[0] = a.x; r[1] = a.y; r[2] = a.z; r[3] = a.w;
  r[4] = b.x; r[5] = b.y; r[6] = b.z; r[7] = b.w;
  return r;
}

// ---------------------------------------------------------------------------
// Kernel 1: qkv = x @ c_attn_w + b  -> Q[B,H,S,HS], K[B,H,S,HS], Vt[B,H,HS,S] (bf16)
// Block tile 128(M) x 128(N), 8 waves, wave tile 64x32, K-step 32.
// ---------------------------------------------------------------------------
__global__ __launch_bounds__(256)
void qkv_kernel(const float* __restrict__ x, const float* __restrict__ w,
                const float* __restrict__ bias,
                bf16_t* __restrict__ Qb, bf16_t* __restrict__ Kb,
                bf16_t* __restrict__ Vt)
{
  __shared__ bf16_t ldsb[128 * 40];   // W tile transposed: [n][k], pad stride 40
  const int t      = threadIdx.x;
  const int wid    = t >> 5;
  const int lane   = t & 31;
  const int lane15 = lane & 15;
  const int kh     = lane >> 4;
  const int wm     = wid & 1;         // 2 row groups of 64
  const int wn     = wid >> 1;        // 4 col groups of 32
  const int cBase  = blockIdx.x * 128;
  const int rBase  = blockIdx.y * 128;

  v8f acc[4][2];
#pragma unroll
  for (int mi = 0; mi < 4; ++mi)
#pragma unroll
    for (int ni = 0; ni < 2; ++ni) acc[mi][ni] = v8f_zero();

  for (int kk = 0; kk < DD; kk += 32) {
    __syncthreads();
    // stage W tile (32k x 128n fp32) transposed to LDS as bf16
#pragma unroll
    for (int i = 0; i < 4; ++i) {
      int idx4 = i * 256 + t;               // 0..1023 float4s
      int krow = idx4 >> 5;                 // 0..31
      int nc   = (idx4 & 31) << 2;          // 0..124
      const float4 wv = *(const float4*)&w[(size_t)(kk + krow) * (3 * DD) + cBase + nc];
      ldsb[(nc + 0) * 40 + krow] = f2bf(wv.x);
      ldsb[(nc + 1) * 40 + krow] = f2bf(wv.y);
      ldsb[(nc + 2) * 40 + krow] = f2bf(wv.z);
      ldsb[(nc + 3) * 40 + krow] = f2bf(wv.w);
    }
    __syncthreads();

    // A fragments straight from global (fp32 -> bf16, packed converts)
    BF16Frag afrag[4];
#pragma unroll
    for (int mi = 0; mi < 4; ++mi) {
      int row = rBase + wm * 64 + mi * 16 + lane15;
      const float* xp = &x[(size_t)row * DD + kk];
      float4 f0 = *(const float4*)(xp + kh * 8);
      float4 f1 = *(const float4*)(xp + kh * 8 + 4);
      float4 f2 = *(const float4*)(xp + 16 + kh * 8);
      float4 f3 = *(const float4*)(xp + 16 + kh * 8 + 4);
      afrag[mi].h[0] = cvt8(pack8(f0, f1));
      afrag[mi].h[1] = cvt8(pack8(f2, f3));
    }

    // B fragments from LDS (per-lane column, contiguous ds_load_b128)
    BF16Frag bfrag[2];
#pragma unroll
    for (int ni = 0; ni < 2; ++ni) {
      int nrow = wn * 32 + ni * 16 + lane15;
      const bf16_t* bp = &ldsb[nrow * 40 + kh * 8];
      bfrag[ni].h[0] = *(const v8bf*)bp;
      bfrag[ni].h[1] = *(const v8bf*)(bp + 16);
    }

#pragma unroll
    for (int mi = 0; mi < 4; ++mi)
#pragma unroll
      for (int ni = 0; ni < 2; ++ni)
        acc[mi][ni] = wmma_bf16(afrag[mi], bfrag[ni], acc[mi][ni]);
  }

  // epilogue: bias + scatter to Q / K / Vt (bf16)
#pragma unroll
  for (int mi = 0; mi < 4; ++mi)
#pragma unroll
    for (int ni = 0; ni < 2; ++ni) {
      int col = cBase + wn * 32 + ni * 16 + lane15;
      int seg = col >> 10;                 // 0=Q, 1=K, 2=V
      int c   = col & (DD - 1);
      int h   = c >> 6, hs = c & 63;
      float bv = bias[col];
#pragma unroll
      for (int i = 0; i < 8; ++i) {
        int row = rBase + wm * 64 + mi * 16 + kh * 8 + i;
        int b   = row >> 11;
        int s   = row & (SS - 1);
        bf16_t o = f2bf(acc[mi][ni][i] + bv);
        if (seg == 0)
          Qb[(((size_t)(b * HH + h) * SS + s) * HSZ) + hs] = o;
        else if (seg == 1)
          Kb[(((size_t)(b * HH + h) * SS + s) * HSZ) + hs] = o;
        else
          Vt[(((size_t)(b * HH + h) * HSZ + hs) * SS) + s] = o;
      }
    }
}

// ---------------------------------------------------------------------------
// Kernel 2: causal flash attention. One wave per 16-query tile, 32-key blocks,
// online softmax; P relayout C->A through per-wave LDS; PV against Vt.
// ---------------------------------------------------------------------------
__global__ __launch_bounds__(128)
void attn_kernel(const bf16_t* __restrict__ Qb, const bf16_t* __restrict__ Kb,
                 const bf16_t* __restrict__ Vt, bf16_t* __restrict__ attn)
{
  __shared__ bf16_t ldsp[4 * 16 * 40];     // per-wave 16x32 P tile, pad stride 40
  const int t      = threadIdx.x;
  const int wid    = t >> 5;
  const int lane   = t & 31;
  const int lane15 = lane & 15;
  const int kh     = lane >> 4;
  const int h      = blockIdx.y;
  const int b      = blockIdx.z;
  const int bh     = b * HH + h;
  const int qt     = blockIdx.x * 4 + wid;
  const int qbase  = qt * 16;
  bf16_t* lP = &ldsp[wid * 16 * 40];

  const bf16_t* Q = Qb + (size_t)bh * SS * HSZ;
  const bf16_t* K = Kb + (size_t)bh * SS * HSZ;
  const bf16_t* V = Vt + (size_t)bh * HSZ * SS;

  // Q fragments (16 queries x 64 hs -> two 16x32 A frags)
  BF16Frag aq[2];
  {
    const bf16_t* qp = Q + (size_t)(qbase + lane15) * HSZ;
    aq[0].h[0] = *(const v8bf*)(qp + kh * 16);
    aq[0].h[1] = *(const v8bf*)(qp + kh * 16 + 8);
    aq[1].h[0] = *(const v8bf*)(qp + 32 + kh * 16);
    aq[1].h[1] = *(const v8bf*)(qp + 32 + kh * 16 + 8);
  }

  v8f O[4];
#pragma unroll
  for (int nt = 0; nt < 4; ++nt) O[nt] = v8f_zero();
  float mrow[8], lrow[8];
#pragma unroll
  for (int i = 0; i < 8; ++i) { mrow[i] = -1e30f; lrow[i] = 0.0f; }

  const float scale = 0.125f;              // 1/sqrt(64)
  const int nkb = (qt + 2) >> 1;           // ceil(16*(qt+1)/32)
  for (int kb = 0; kb < nkb; ++kb) {
    const int kbase = kb * 32;

    // scores: two 16x16 tiles (keys kbase.. and kbase+16..)
    v8f s0 = v8f_zero(), s1 = v8f_zero();
    {
      const bf16_t* kpa = K + (size_t)(kbase + lane15) * HSZ;
      const bf16_t* kpb = K + (size_t)(kbase + 16 + lane15) * HSZ;
      BF16Frag bk;
      bk.h[0] = *(const v8bf*)(kpa + kh * 16);
      bk.h[1] = *(const v8bf*)(kpa + kh * 16 + 8);
      s0 = wmma_bf16(aq[0], bk, s0);
      bk.h[0] = *(const v8bf*)(kpa + 32 + kh * 16);
      bk.h[1] = *(const v8bf*)(kpa + 32 + kh * 16 + 8);
      s0 = wmma_bf16(aq[1], bk, s0);
      bk.h[0] = *(const v8bf*)(kpb + kh * 16);
      bk.h[1] = *(const v8bf*)(kpb + kh * 16 + 8);
      s1 = wmma_bf16(aq[0], bk, s1);
      bk.h[0] = *(const v8bf*)(kpb + 32 + kh * 16);
      bk.h[1] = *(const v8bf*)(kpb + 32 + kh * 16 + 8);
      s1 = wmma_bf16(aq[1], bk, s1);
    }

    const int key0 = kbase + lane15;
    const int key1 = key0 + 16;
#pragma unroll
    for (int i = 0; i < 8; ++i) {
      int q = qbase + kh * 8 + i;
      float a0 = (key0 <= q) ? s0[i] * scale : -1e30f;
      float a1 = (key1 <= q) ? s1[i] * scale : -1e30f;
      float rm = fmaxf(a0, a1);
#pragma unroll
      for (int off = 1; off < 16; off <<= 1)
        rm = fmaxf(rm, __shfl_xor(rm, off, 32));
      float mnew  = fmaxf(mrow[i], rm);
      float alpha = __expf(mrow[i] - mnew);
      float p0 = __expf(a0 - mnew);
      float p1 = __expf(a1 - mnew);
      float rs = p0 + p1;
#pragma unroll
      for (int off = 1; off < 16; off <<= 1)
        rs += __shfl_xor(rs, off, 32);
      lrow[i] = lrow[i] * alpha + rs;
      mrow[i] = mnew;
      O[0][i] *= alpha; O[1][i] *= alpha; O[2][i] *= alpha; O[3][i] *= alpha;
      // stage P (C layout -> row-major LDS) for A-fragment reload
      int m = kh * 8 + i;
      lP[m * 40 + lane15]      = f2bf(p0);
      lP[m * 40 + 16 + lane15] = f2bf(p1);
    }
    asm volatile("s_wait_dscnt 0" ::: "memory");

    BF16Frag pf;
    {
      const bf16_t* pp = &lP[lane15 * 40 + kh * 16];
      pf.h[0] = *(const v8bf*)pp;
      pf.h[1] = *(const v8bf*)(pp + 8);
    }
#pragma unroll
    for (int nt = 0; nt < 4; ++nt) {
      const bf16_t* vp = V + (size_t)(nt * 16 + lane15) * SS + kbase + kh * 16;
      BF16Frag vf;
      vf.h[0] = *(const v8bf*)vp;
      vf.h[1] = *(const v8bf*)(vp + 8);
      O[nt] = wmma_bf16(pf, vf, O[nt]);
    }
  }

  // normalize + store attn[b, s, h*64 + hs] as bf16
#pragma unroll
  for (int nt = 0; nt < 4; ++nt)
#pragma unroll
    for (int i = 0; i < 8; ++i) {
      int s   = qbase + kh * 8 + i;
      int col = h * HSZ + nt * 16 + lane15;
      attn[(size_t)(b * SS + s) * DD + col] = f2bf(O[nt][i] / lrow[i]);
    }
}

// ---------------------------------------------------------------------------
// Kernel 3: out = attn @ c_proj_w + b  (bf16 A, fp32 out). Same tiling as k1.
// ---------------------------------------------------------------------------
__global__ __launch_bounds__(256)
void proj_kernel(const bf16_t* __restrict__ attn, const float* __restrict__ w,
                 const float* __restrict__ bias, float* __restrict__ out)
{
  __shared__ bf16_t ldsb[128 * 40];
  const int t      = threadIdx.x;
  const int wid    = t >> 5;
  const int lane   = t & 31;
  const int lane15 = lane & 15;
  const int kh     = lane >> 4;
  const int wm     = wid & 1;
  const int wn     = wid >> 1;
  const int cBase  = blockIdx.x * 128;
  const int rBase  = blockIdx.y * 128;

  v8f acc[4][2];
#pragma unroll
  for (int mi = 0; mi < 4; ++mi)
#pragma unroll
    for (int ni = 0; ni < 2; ++ni) acc[mi][ni] = v8f_zero();

  for (int kk = 0; kk < DD; kk += 32) {
    __syncthreads();
#pragma unroll
    for (int i = 0; i < 4; ++i) {
      int idx4 = i * 256 + t;
      int krow = idx4 >> 5;
      int nc   = (idx4 & 31) << 2;
      const float4 wv = *(const float4*)&w[(size_t)(kk + krow) * DD + cBase + nc];
      ldsb[(nc + 0) * 40 + krow] = f2bf(wv.x);
      ldsb[(nc + 1) * 40 + krow] = f2bf(wv.y);
      ldsb[(nc + 2) * 40 + krow] = f2bf(wv.z);
      ldsb[(nc + 3) * 40 + krow] = f2bf(wv.w);
    }
    __syncthreads();

    BF16Frag afrag[4];
#pragma unroll
    for (int mi = 0; mi < 4; ++mi) {
      int row = rBase + wm * 64 + mi * 16 + lane15;
      const bf16_t* ap = attn + (size_t)row * DD + kk;
      afrag[mi].h[0] = *(const v8bf*)(ap + kh * 8);
      afrag[mi].h[1] = *(const v8bf*)(ap + 16 + kh * 8);
    }
    BF16Frag bfrag[2];
#pragma unroll
    for (int ni = 0; ni < 2; ++ni) {
      int nrow = wn * 32 + ni * 16 + lane15;
      const bf16_t* bp = &ldsb[nrow * 40 + kh * 8];
      bfrag[ni].h[0] = *(const v8bf*)bp;
      bfrag[ni].h[1] = *(const v8bf*)(bp + 16);
    }
#pragma unroll
    for (int mi = 0; mi < 4; ++mi)
#pragma unroll
      for (int ni = 0; ni < 2; ++ni)
        acc[mi][ni] = wmma_bf16(afrag[mi], bfrag[ni], acc[mi][ni]);
  }

#pragma unroll
  for (int mi = 0; mi < 4; ++mi)
#pragma unroll
    for (int ni = 0; ni < 2; ++ni) {
      int col = cBase + wn * 32 + ni * 16 + lane15;
      float bv = bias[col];
#pragma unroll
      for (int i = 0; i < 8; ++i) {
        int row = rBase + wm * 64 + mi * 16 + kh * 8 + i;
        out[(size_t)row * DD + col] = acc[mi][ni][i] + bv;
      }
    }
}

// ---------------------------------------------------------------------------
extern "C" void kernel_launch(void* const* d_in, const int* in_sizes, int n_in,
                              void* d_out, int out_size, void* d_ws, size_t ws_size,
                              hipStream_t stream) {
  (void)in_sizes; (void)n_in; (void)out_size; (void)ws_size;
  const float* x        = (const float*)d_in[0];
  const float* c_attn_w = (const float*)d_in[1];
  const float* c_attn_b = (const float*)d_in[2];
  const float* c_proj_w = (const float*)d_in[3];
  const float* c_proj_b = (const float*)d_in[4];
  float* out = (float*)d_out;

  const size_t headElems = (size_t)BB * HH * SS * HSZ;   // 4M elems
  bf16_t* Qb   = (bf16_t*)d_ws;
  bf16_t* Kb   = Qb + headElems;
  bf16_t* Vt   = Kb + headElems;
  bf16_t* attn = Vt + headElems;                          // BB*SS*DD elems

  qkv_kernel<<<dim3((3 * DD) / 128, (BB * SS) / 128), 256, 0, stream>>>(
      x, c_attn_w, c_attn_b, Qb, Kb, Vt);
  attn_kernel<<<dim3(SS / 64, HH, BB), 128, 0, stream>>>(Qb, Kb, Vt, attn);
  proj_kernel<<<dim3(DD / 128, (BB * SS) / 128), 256, 0, stream>>>(
      attn, c_proj_w, c_proj_b, out);
}
